// SwinBlock_12421045420429
// MI455X (gfx1250) — matrix operand order repaired and checked by reading, not compiled, not run on hardware
//
#include <hip/hip_runtime.h>
#include <hip/hip_bf16.h>

// ---------------------------------------------------------------------------
// Types for CDNA5 WMMA (wave32)
// ---------------------------------------------------------------------------
typedef __bf16 v16bf __attribute__((ext_vector_type(16)));
typedef float  v8f   __attribute__((ext_vector_type(8)));
typedef unsigned short u16;

// fp32 -> bf16 round-to-nearest-even
static __device__ __forceinline__ u16 f2bf(float f) {
    unsigned u = __float_as_uint(f);
    u += 0x7fffu + ((u >> 16) & 1u);
    return (u16)(u >> 16);
}

static __device__ __forceinline__ v8f wmma_bf16(v16bf a, v16bf b, v8f c) {
    return __builtin_amdgcn_wmma_f32_16x16x32_bf16(
        /*neg_a=*/false, a, /*neg_b=*/false, b,
        /*c_mod=*/(short)0, c, /*reuse_a=*/false, /*reuse_b=*/false);
}

// A fragment: 16x32 bf16 (M x K).  lane&15 = M row; half-wave selects K group.
// VGPR layout: lane<16 holds K {0..15}, lane>=16 {16..31} (matches frag_b)
static __device__ __forceinline__ v16bf frag_a(const u16* lds, int ldm, int row0, int lane) {
    int m  = row0 + (lane & 15);
    int kb = (lane >> 4) << 4;          // 0 or 16
    v16bf a;
#pragma unroll
    for (int e = 0; e < 16; ++e) {
        int k = kb + e;
        a[e] = __builtin_bit_cast(__bf16, lds[m * ldm + k]);
    }
    return a;
}

// B fragment: 32x16 bf16 (K x N).  lane&15 = N col; half-wave selects K half.
// element e holds K = (lane>=16?16:0)+e.  Addressing B[k][n] = lds[k*sK + n*sN].
static __device__ __forceinline__ v16bf frag_b(const u16* lds, int sK, int sN, int col0, int lane) {
    int n  = col0 + (lane & 15);
    int kb = (lane >> 4) << 4;          // 0 or 16
    v16bf b;
#pragma unroll
    for (int e = 0; e < 16; ++e)
        b[e] = __builtin_bit_cast(__bf16, lds[(kb + e) * sK + n * sN]);
    return b;
}

// Async global -> LDS copy of 16 bytes (CDNA5, ASYNCcnt-tracked).
static __device__ __forceinline__ void async_copy_b128(const u16* gptr, const u16* lptr) {
    unsigned lds_off = (unsigned)(size_t)lptr;   // generic LDS addr low 32b = LDS byte offset
    asm volatile("global_load_async_to_lds_b128 %0, %1, off"
                 :: "v"(lds_off), "v"(gptr) : "memory");
}

// ---------------------------------------------------------------------------
// f32 -> bf16 conversion (weights)
// ---------------------------------------------------------------------------
__global__ void f32_to_bf16_kernel(const float* __restrict__ src, u16* __restrict__ dst, int n) {
    int i = blockIdx.x * blockDim.x + threadIdx.x;
    if (i < n) dst[i] = f2bf(src[i]);
}

// ---------------------------------------------------------------------------
// Fused LayerNorm (C=192) -> bf16.  One wave32 per row, 6 elems per lane.
// ---------------------------------------------------------------------------
__global__ __launch_bounds__(256) void ln_bf16_kernel(
    const float* __restrict__ x, const float* __restrict__ g,
    const float* __restrict__ bta, u16* __restrict__ out, int rows) {
    int wave = (blockIdx.x * blockDim.x + threadIdx.x) >> 5;
    int lane = threadIdx.x & 31;
    if (wave >= rows) return;
    const float* xr = x + (size_t)wave * 192;
    float v[6], s = 0.f, s2 = 0.f;
#pragma unroll
    for (int i = 0; i < 6; ++i) {
        v[i] = xr[lane + i * 32];
        s += v[i];
        s2 += v[i] * v[i];
    }
#pragma unroll
    for (int off = 16; off > 0; off >>= 1) {
        s  += __shfl_xor(s,  off, 32);
        s2 += __shfl_xor(s2, off, 32);
    }
    float mu  = s * (1.0f / 192.0f);
    float var = s2 * (1.0f / 192.0f) - mu * mu;
    float inv = rsqrtf(var + 1e-5f);
    u16* orow = out + (size_t)wave * 192;
#pragma unroll
    for (int i = 0; i < 6; ++i) {
        int c = lane + i * 32;
        orow[c] = f2bf((v[i] - mu) * inv * g[c] + bta[c]);
    }
}

// ---------------------------------------------------------------------------
// Tiled WMMA GEMM with double-buffered ASYNC global->LDS tile copies.
//   C[M,N] = A[M,K](bf16) @ B[K,N](bf16) + bias, epilogues.
// Block tile 128x64, BK=32, 256 threads = 8 waves (4x2), wave tile 32x32.
// Per K-step each thread issues 3 async b128 copies (2 for A, 1 for B);
// s_wait_asynccnt 3 retires the previous batch while the next one flies.
// EPI: 0 = bias -> bf16 out;  1 = bias+GELU -> bf16 out;
//      2 = bias + residual(f32) -> f32 out.
// M%128==0, N%64==0, K%32==0 required.
// ---------------------------------------------------------------------------
template <int EPI>
__global__ __launch_bounds__(256) void gemm_bf16_kernel(
    const u16* __restrict__ A, const u16* __restrict__ B,
    const float* __restrict__ bias, const float* __restrict__ res,
    void* __restrict__ out, int M, int N, int K) {
    // Row strides padded so every 16B async chunk is 16B-aligned in LDS.
    __shared__ u16 As[2][128 * 40];   // 128 rows x 32 cols (pad 8): 10 KB each
    __shared__ u16 Bs[2][32 * 72];    // 32 rows x 64 cols (pad 8): 4.5 KB each

    const int bm = blockIdx.y * 128;
    const int bn = blockIdx.x * 64;
    const int t = threadIdx.x;
    const int lane = t & 31;
    const int wave = t >> 5;
    const int wm = wave >> 1;     // 0..3
    const int wn = wave & 1;      // 0..1

    auto issue_tile = [&](int buf, int k0) {
        // A tile: 128x32 bf16 = 512 x b128 chunks; 2 per thread.
#pragma unroll
        for (int i = 0; i < 2; ++i) {
            int pos = (i * 256 + t) * 8;       // element index in tile
            int r = pos >> 5, c = pos & 31;
            async_copy_b128(&A[(size_t)(bm + r) * K + k0 + c], &As[buf][r * 40 + c]);
        }
        // B tile: 32x64 bf16 = 256 x b128 chunks; 1 per thread.
        {
            int pos = t * 8;
            int r = pos >> 6, c = pos & 63;
            async_copy_b128(&B[(size_t)(k0 + r) * N + bn + c], &Bs[buf][r * 72 + c]);
        }
    };

    v8f acc[2][2] = {};

    issue_tile(0, 0);
    int buf = 0;
    for (int k0 = 0; k0 < K; k0 += 32) {
        if (k0 + 32 < K) {
            issue_tile(buf ^ 1, k0 + 32);                     // prefetch next tile
            asm volatile("s_wait_asynccnt 0x3" ::: "memory"); // current tile done
        } else {
            asm volatile("s_wait_asynccnt 0x0" ::: "memory");
        }
        __syncthreads();

        v16bf a0 = frag_a(As[buf], 40, wm * 32, lane);
        v16bf a1 = frag_a(As[buf], 40, wm * 32 + 16, lane);
        v16bf b0 = frag_b(Bs[buf], 72, 1, wn * 32, lane);
        v16bf b1 = frag_b(Bs[buf], 72, 1, wn * 32 + 16, lane);
        acc[0][0] = wmma_bf16(a0, b0, acc[0][0]);
        acc[0][1] = wmma_bf16(a0, b1, acc[0][1]);
        acc[1][0] = wmma_bf16(a1, b0, acc[1][0]);
        acc[1][1] = wmma_bf16(a1, b1, acc[1][1]);
        __syncthreads();   // protect this buffer from the next async overwrite
        buf ^= 1;
    }

    // Epilogue.  C/D layout: lane&15 = N col; elem e -> M row e + (lane>=16?8:0).
    const int mo = (lane >> 4) << 3;
#pragma unroll
    for (int mi = 0; mi < 2; ++mi) {
#pragma unroll
        for (int ni = 0; ni < 2; ++ni) {
            int col = bn + wn * 32 + ni * 16 + (lane & 15);
            float bv = bias[col];
#pragma unroll
            for (int e = 0; e < 8; ++e) {
                int row = bm + wm * 32 + mi * 16 + mo + e;
                float val = acc[mi][ni][e] + bv;
                if (EPI == 1) {  // exact GELU
                    val = 0.5f * val * (1.0f + erff(val * 0.70710678118654752f));
                }
                if (EPI == 2) {
                    val += res[(size_t)row * N + col];
                    ((float*)out)[(size_t)row * N + col] = val;
                } else {
                    ((u16*)out)[(size_t)row * N + col] = f2bf(val);
                }
            }
        }
    }
}

// ---------------------------------------------------------------------------
// Windowed shifted attention: one wave per (window, head).
// N=49 padded to 64, hd=32.  S = Q K^T (16 WMMA), softmax(+rpb+mask), P V (16 WMMA).
// Shift roll folded into gather/scatter indices (same index both ways).
// ---------------------------------------------------------------------------
__global__ __launch_bounds__(32) void attn_kernel(
    const u16* __restrict__ qkv,       // (B*3136, 576) bf16
    const float* __restrict__ rpb,     // (169, 6) f32
    u16* __restrict__ out) {           // (B*3136, 192) bf16
    __shared__ u16  Qs[64 * 32];
    __shared__ u16  Ks[64 * 32];
    __shared__ u16  Vs[64 * 32];
    __shared__ float Ss[64 * 64];
    __shared__ u16  Ps[64 * 64];
    __shared__ float biasT[169];
    __shared__ int  growA[64];
    __shared__ int  regA[64];

    const int lane = threadIdx.x & 31;
    int bid = blockIdx.x;
    const int head = bid % 6;  bid /= 6;
    const int w = bid % 64;    const int b = bid / 64;
    const int wr = w >> 3, wc = w & 7;

    // Per-token global row index (shift-rolled) + region id for the mask.
    for (int tkn = lane; tkn < 64; tkn += 32) {
        if (tkn < 49) {
            int tr = tkn / 7, tc = tkn % 7;
            int si = wr * 7 + tr, sj = wc * 7 + tc;         // shifted coords
            int ii = (si + 3) % 56, jj = (sj + 3) % 56;     // source == dest row
            growA[tkn] = b * 3136 + ii * 56 + jj;
            int rh = (si < 49) ? 0 : ((si < 53) ? 1 : 2);
            int rw = (sj < 49) ? 0 : ((sj < 53) ? 1 : 2);
            regA[tkn] = rh * 3 + rw;
        } else {
            growA[tkn] = -1;
            regA[tkn] = 0;
        }
    }
    for (int i = lane; i < 169; i += 32) biasT[i] = rpb[i * 6 + head];
    __syncthreads();

    // Gather Q/K/V (head slice) into LDS, zero-padding rows 49..63.
    for (int r = 0; r < 64; ++r) {
        int gr = growA[r];
        u16 qv = 0, kv = 0, vv = 0;
        if (gr >= 0) {
            size_t base = (size_t)gr * 576 + head * 32 + lane;
            qv = qkv[base];
            kv = qkv[base + 192];
            vv = qkv[base + 384];
        }
        Qs[r * 32 + lane] = qv;
        Ks[r * 32 + lane] = kv;
        Vs[r * 32 + lane] = vv;
    }
    __syncthreads();

    // S = Q @ K^T : 4x4 tiles, K-dim = 32 = one wmma step.
    const int mo = (lane >> 4) << 3;
#pragma unroll
    for (int mt = 0; mt < 4; ++mt) {
        v16bf a = frag_a(Qs, 32, mt * 16, lane);
#pragma unroll
        for (int nt = 0; nt < 4; ++nt) {
            v16bf bb = frag_b(Ks, /*sK=*/1, /*sN=*/32, nt * 16, lane);  // K^T
            v8f s = {};
            s = wmma_bf16(a, bb, s);
#pragma unroll
            for (int e = 0; e < 8; ++e)
                Ss[(mt * 16 + mo + e) * 64 + nt * 16 + (lane & 15)] = s[e];
        }
    }
    __syncthreads();

    // Softmax with scale, relative position bias, shift mask.
    const float scale = 0.1767766952966369f;  // 32^-0.5
    for (int rr = lane; rr < 64; rr += 32) {
        if (rr < 49) {
            int tr1 = rr / 7, tc1 = rr % 7, rg1 = regA[rr];
            float mx = -1e30f;
            for (int m = 0; m < 49; ++m) {
                int tr2 = m / 7, tc2 = m % 7;
                float bv = biasT[(tr1 - tr2 + 6) * 13 + (tc1 - tc2 + 6)];
                float mk = (rg1 != regA[m]) ? -100.0f : 0.0f;
                float v = Ss[rr * 64 + m] * scale + bv + mk;
                Ss[rr * 64 + m] = v;
                mx = fmaxf(mx, v);
            }
            float sum = 0.f;
            for (int m = 0; m < 49; ++m) {
                float e = __expf(Ss[rr * 64 + m] - mx);
                Ss[rr * 64 + m] = e;
                sum += e;
            }
            float inv = 1.0f / sum;
            for (int m = 0; m < 49; ++m) Ps[rr * 64 + m] = f2bf(Ss[rr * 64 + m] * inv);
            for (int m = 49; m < 64; ++m) Ps[rr * 64 + m] = 0;
        } else {
            for (int m = 0; m < 64; ++m) Ps[rr * 64 + m] = 0;
        }
    }
    __syncthreads();

    // O = P @ V : 4x2 tiles, K=64 -> two wmma steps. Scatter with inverse roll.
#pragma unroll
    for (int mt = 0; mt < 4; ++mt) {
#pragma unroll
        for (int nt = 0; nt < 2; ++nt) {
            v8f o = {};
#pragma unroll
            for (int kk = 0; kk < 64; kk += 32) {
                v16bf a = frag_a(Ps + kk, 64, mt * 16, lane);
                v16bf bb = frag_b(Vs + kk * 32, /*sK=*/32, /*sN=*/1, nt * 16, lane);
                o = wmma_bf16(a, bb, o);
            }
#pragma unroll
            for (int e = 0; e < 8; ++e) {
                int tkn = mt * 16 + mo + e;
                if (tkn < 49) {
                    int gr = growA[tkn];
                    int col = head * 32 + nt * 16 + (lane & 15);
                    out[(size_t)gr * 192 + col] = f2bf(o[e]);
                }
            }
        }
    }
}

// ---------------------------------------------------------------------------
// Host-side orchestration
// ---------------------------------------------------------------------------
extern "C" void kernel_launch(void* const* d_in, const int* in_sizes, int n_in,
                              void* d_out, int out_size, void* d_ws, size_t ws_size,
                              hipStream_t stream) {
    const float* x      = (const float*)d_in[0];
    const float* n1g    = (const float*)d_in[1];
    const float* n1b    = (const float*)d_in[2];
    const float* n2g    = (const float*)d_in[3];
    const float* n2b    = (const float*)d_in[4];
    const float* qkv_w  = (const float*)d_in[5];
    const float* qkv_b  = (const float*)d_in[6];
    const float* proj_w = (const float*)d_in[7];
    const float* proj_b = (const float*)d_in[8];
    const float* rpb    = (const float*)d_in[9];
    const float* mlp_w1 = (const float*)d_in[10];
    const float* mlp_b1 = (const float*)d_in[11];
    const float* mlp_w2 = (const float*)d_in[12];
    const float* mlp_b2 = (const float*)d_in[13];

    const int B = 32, HW = 3136, C = 192;
    const int M = B * HW;            // 100352
    (void)in_sizes; (void)n_in; (void)out_size; (void)ws_size;

    // Workspace carve-out (256B aligned).
    char* base = (char*)d_ws;
    size_t off = 0;
    auto alloc = [&](size_t bytes) -> void* {
        off = (off + 255) & ~(size_t)255;
        void* p = base + off;
        off += bytes;
        return p;
    };
    u16* w_qkv  = (u16*)alloc((size_t)C * 3 * C * 2);          // 192x576
    u16* w_proj = (u16*)alloc((size_t)C * C * 2);              // 192x192
    u16* w_m1   = (u16*)alloc((size_t)C * 4 * C * 2);          // 192x768
    u16* w_m2   = (u16*)alloc((size_t)4 * C * C * 2);          // 768x192
    u16* hn     = (u16*)alloc((size_t)M * C * 2);              // ln1 out (bf16)
    u16* qkvb   = (u16*)alloc((size_t)M * 3 * C * 2);          // qkv (bf16)
    u16* attn_o = (u16*)alloc((size_t)M * C * 2);              // attn out (bf16)
    float* hbuf = (float*)alloc((size_t)M * C * 4);            // h = proj + x (f32)
    u16* ln2    = (u16*)alloc((size_t)M * C * 2);              // ln2 out (bf16)
    u16* mid    = (u16*)alloc((size_t)M * 4 * C * 2);          // gelu(mlp1) (bf16)

    auto cvt = [&](const float* s, u16* d, int n) {
        f32_to_bf16_kernel<<<(n + 255) / 256, 256, 0, stream>>>(s, d, n);
    };
    cvt(qkv_w, w_qkv, C * 3 * C);
    cvt(proj_w, w_proj, C * C);
    cvt(mlp_w1, w_m1, C * 4 * C);
    cvt(mlp_w2, w_m2, 4 * C * C);

    // 1) LN1 -> bf16
    ln_bf16_kernel<<<M / 8, 256, 0, stream>>>(x, n1g, n1b, hn, M);

    // 2) qkv GEMM: (M,192)x(192,576) -> bf16
    gemm_bf16_kernel<0><<<dim3((3 * C) / 64, M / 128), 256, 0, stream>>>(
        hn, w_qkv, qkv_b, nullptr, qkvb, M, 3 * C, C);

    // 3) windowed shifted attention (2048 windows x 6 heads)
    attn_kernel<<<B * 64 * 6, 32, 0, stream>>>(qkvb, rpb, attn_o);

    // 4) proj GEMM + residual(x): -> h (f32)
    gemm_bf16_kernel<2><<<dim3(C / 64, M / 128), 256, 0, stream>>>(
        attn_o, w_proj, proj_b, x, hbuf, M, C, C);

    // 5) LN2 -> bf16
    ln_bf16_kernel<<<M / 8, 256, 0, stream>>>(hbuf, n2g, n2b, ln2, M);

    // 6) MLP1 GEMM + GELU: (M,192)x(192,768) -> bf16
    gemm_bf16_kernel<1><<<dim3((4 * C) / 64, M / 128), 256, 0, stream>>>(
        ln2, w_m1, mlp_b1, nullptr, mid, M, 4 * C, C);

    // 7) MLP2 GEMM + residual(h): (M,768)x(768,192) -> d_out (f32)
    gemm_bf16_kernel<2><<<dim3(C / 64, M / 128), 256, 0, stream>>>(
        mid, w_m2, mlp_b2, hbuf, (float*)d_out, M, C, 4 * C);
}
